// CAER_61624190763532
// MI455X (gfx1250) — compile-verified, matrix operand order checked by reading, not run on hardware
//
#include <hip/hip_runtime.h>

typedef __attribute__((ext_vector_type(2)))  float    v2f;
typedef __attribute__((ext_vector_type(4)))  float    v4f;
typedef __attribute__((ext_vector_type(8)))  float    v8f;
typedef __attribute__((ext_vector_type(16))) _Float16 v16h;

#define B_  64
#define S_  2048
#define D_  300
#define SA_ 8

// ---------------------------------------------------------------------------
// Kernel 1: u1[b,s] = embed[idx[b,s]] . w + bias  via WMMA (16 tokens / wave),
// plus per-batch masked sum & count (one atomicAdd pair per wave).
// Block = 128 threads = 4 waves = 64 tokens; 32 blocks per batch row b.
// ---------------------------------------------------------------------------
__global__ __launch_bounds__(128)
void CAER_u_kernel(const int* __restrict__ text_idx,
                   const float* __restrict__ embed,
                   const float* __restrict__ fc_w,
                   const float* __restrict__ fc_b,
                   float* __restrict__ u_out,
                   float* __restrict__ sum_b,
                   unsigned* __restrict__ cnt_b)
{
    const int wave = threadIdx.x >> 5;      // 0..3
    const int lane = threadIdx.x & 31;
    const int b    = blockIdx.x >> 5;       // 32 blocks per b (2048/64)
    const int s0   = (blockIdx.x & 31) * 64 + wave * 16;  // this wave's 16 tokens
    const int tok  = lane & 15;             // lanes l and l+16 share a token row

    const int idx = text_idx[b * S_ + s0 + tok];
    const float* __restrict__ row = embed + (size_t)idx * D_;

    v8f c = {};

#if __has_builtin(__builtin_amdgcn_wmma_f32_16x16x4_f32)
    // f32 WMMA: A 16x4 (v2f/lane: lanes 0-15 K={0,1}, lanes 16-31 K={2,3});
    // B 4x16 with every column = w chunk (same per-lane K indexing as A).
    const int koff = (lane >> 4) * 2;       // 0 or 2
    #pragma unroll 5
    for (int k0 = 0; k0 < D_; k0 += 4) {    // 300/4 = 75 exact, no tail
        const int k = k0 + koff;
        v2f a; a.x = row[k];  a.y = row[k + 1];
        v2f w; w.x = fc_w[k]; w.y = fc_w[k + 1];
        c = __builtin_amdgcn_wmma_f32_16x16x4_f32(
                /*neg_a=*/false, a, /*neg_b=*/false, w,
                /*c_mod=*/(short)0, c, /*reuse_a=*/false, /*reuse_b=*/false);
    }
#else
    // Fallback: codegen-confirmed f16 WMMA, K padded 300 -> 320 with zeros.
    // 16-bit A layout: lane-half h: elems 0..7 = K[h*8 .. h*8+7],
    // elems 8..15 = K[16+h*8 .. 16+h*8+7]; B mirrors in K (w depends on K only).
    const int h8 = (lane >> 4) * 8;
    for (int k0 = 0; k0 < 320; k0 += 32) {
        v16h a, w;
        #pragma unroll
        for (int j = 0; j < 8; ++j) {
            const int kA = k0 + h8 + j;
            const int kB = k0 + 16 + h8 + j;
            a[j]     = (kA < D_) ? (_Float16)row[kA]  : (_Float16)0.f;
            a[8 + j] = (kB < D_) ? (_Float16)row[kB]  : (_Float16)0.f;
            w[j]     = (kA < D_) ? (_Float16)fc_w[kA] : (_Float16)0.f;
            w[8 + j] = (kB < D_) ? (_Float16)fc_w[kB] : (_Float16)0.f;
        }
        c = __builtin_amdgcn_wmma_f32_16x16x32_f16(
                false, a, false, w, (short)0, c, false, false);
    }
#endif

    // C/D layout: VGPR v, lanes 0-15 -> M=v (N=lane), lanes 16-31 -> M=v+8.
    // Every N column holds the same u1, so lane 0 / lane 16 suffice.
    const float bias = fc_b[0];
    float u_lo[8], u_hi[8];
    #pragma unroll
    for (int v = 0; v < 8; ++v) {
        u_lo[v] = __shfl(c[v], 0, 32)  + bias;   // tokens 0..7
        u_hi[v] = __shfl(c[v], 16, 32) + bias;   // tokens 8..15
    }

    // Valid-token mask (idx != 0 <=> positional prefix mask by construction).
    const unsigned long long ball = __ballot(lane < 16 && idx != 0);
    const unsigned mask16 = (unsigned)(ball & 0xFFFFu);

    if (lane == 0) {
        float* __restrict__ up = u_out + b * S_ + s0;
        float local = 0.f;
        #pragma unroll
        for (int v = 0; v < 8; ++v) {
            up[v]     = u_lo[v];
            up[8 + v] = u_hi[v];
            if (mask16 & (1u << v))       local += u_lo[v];
            if (mask16 & (1u << (8 + v))) local += u_hi[v];
        }
        atomicAdd(&sum_b[b], local);
        atomicAdd(&cnt_b[b], (unsigned)__popc(mask16));
    }
}

// ---------------------------------------------------------------------------
// Kernel 2: output[b,s,:] = embed[idx] * ((idx!=0 && u1>=mean) ? u1 : 0).
// One wave per token, coalesced float4 traffic (embed rows are L2-resident).
// ---------------------------------------------------------------------------
__global__ __launch_bounds__(256)
void CAER_out_kernel(const int* __restrict__ text_idx,
                     const float* __restrict__ embed,
                     const float* __restrict__ u_in,
                     const float* __restrict__ sum_b,
                     const unsigned* __restrict__ cnt_b,
                     float* __restrict__ out)
{
    const int wave = threadIdx.x >> 5;                 // 8 tokens per block
    const int lane = threadIdx.x & 31;
    const long long t = (long long)blockIdx.x * 8 + wave;   // token in [0, B*S)
    const int b   = (int)(t / S_);
    const int idx = text_idx[t];
    const float u    = u_in[t];
    const float mean = sum_b[b] / (float)cnt_b[b];     // text_len >= S/4 > 0
    const float scale = (idx != 0 && u >= mean) ? u : 0.0f;

    const float* __restrict__ row  = embed + (size_t)idx * D_;
    float* __restrict__       orow = out + t * D_;
    for (int d = lane * 4; d < D_; d += 128) {         // 300 = 75 float4
        v4f v = *(const v4f*)(row + d);
        v4f r; r.x = v.x * scale; r.y = v.y * scale;
               r.z = v.z * scale; r.w = v.w * scale;
        *(v4f*)(orow + d) = r;
    }
}

// ---------------------------------------------------------------------------
// Kernel 3: orig_target_embed[b,i,:] = embed[target_idx[b,i]]  (512 rows)
// ---------------------------------------------------------------------------
__global__ __launch_bounds__(256)
void CAER_tgt_kernel(const int* __restrict__ tgt_idx,
                     const float* __restrict__ embed,
                     float* __restrict__ out)
{
    const int wave = threadIdx.x >> 5;
    const int lane = threadIdx.x & 31;
    const int t = blockIdx.x * 8 + wave;               // 0..511, grid exact
    const int idx = tgt_idx[t];
    const float* __restrict__ row  = embed + (size_t)idx * D_;
    float* __restrict__       orow = out + (size_t)t * D_;
    for (int d = lane * 4; d < D_; d += 128)
        *(v4f*)(orow + d) = *(const v4f*)(row + d);
}

extern "C" void kernel_launch(void* const* d_in, const int* in_sizes, int n_in,
                              void* d_out, int out_size, void* d_ws, size_t ws_size,
                              hipStream_t stream)
{
    (void)in_sizes; (void)n_in; (void)out_size; (void)ws_size;

    const int*   text_idx = (const int*)d_in[0];
    /* d_in[1] = aspect_indices: unused by the reference outputs */
    const int*   tgt_idx  = (const int*)d_in[2];
    const float* embed    = (const float*)d_in[3];
    const float* fc_w     = (const float*)d_in[4];
    const float* fc_b     = (const float*)d_in[5];
    float*       out      = (float*)d_out;

    // Workspace layout: [u1: B*S f32][sum: B f32][cnt: B u32]
    float*    u_ws  = (float*)d_ws;
    float*    sum_b = u_ws + (size_t)B_ * S_;
    unsigned* cnt_b = (unsigned*)(sum_b + B_);

    hipMemsetAsync(sum_b, 0, B_ * (sizeof(float) + sizeof(unsigned)), stream);

    CAER_u_kernel<<<B_ * (S_ / 64), 128, 0, stream>>>(
        text_idx, embed, fc_w, fc_b, u_ws, sum_b, cnt_b);

    CAER_out_kernel<<<(B_ * S_) / 8, 256, 0, stream>>>(
        text_idx, embed, u_ws, sum_b, cnt_b, out);

    CAER_tgt_kernel<<<(B_ * SA_) / 8, 256, 0, stream>>>(
        tgt_idx, embed, out + (size_t)B_ * S_ * D_);
}
